// TemGNN_50019189129488
// MI455X (gfx1250) — compile-verified, matrix operand order checked by reading, not run on hardware
//
#include <hip/hip_runtime.h>
#include <hip/hip_bf16.h>
#include <math.h>

// ---- problem constants (match reference) ----
#define B_GRAPHS 512
#define NV_PER   40
#define NA_PER   100
#define NVN      (B_GRAPHS * NV_PER)   // 20480 video nodes
#define NAN_     (B_GRAPHS * NA_PER)   // 51200 audio nodes
#define EVV      (NVN * 8)             // 163840
#define EAA      (NAN_ * 8)            // 409600
#define EVA      (NVN * 3)             // 61440
#define DVDIM    1024
#define DADIM    128
#define HDIM     512
#define OUTD     128
#define LAYERS   4
#define EPSF     1e-5f

typedef __bf16 bf16x16 __attribute__((ext_vector_type(16)));
typedef float  f32x8   __attribute__((ext_vector_type(8)));

// Fragment K mapping (CDNA5 ISA 7.12.2, 16-bit A 16x32):
//   lane half = lane>>4 selects kbase = {0,8}; element i -> k = kbase + (i<8 ? i : i+8)

// ---------------------------------------------------------------------------
// Repack activation A[M,K] f32 row-major -> bf16 fragment-major:
//   layout [tileM][kblock][lane 0..31][16 elems]; one wave per (tileM,kblock).
// ---------------------------------------------------------------------------
__global__ __launch_bounds__(256)
void k_repack_a(const float* __restrict__ A, __bf16* __restrict__ out, int M, int K)
{
    const int lane = threadIdx.x & 31;
    const int wid  = blockIdx.x * 8 + (threadIdx.x >> 5);
    const int Kb   = K >> 5;
    const int tiles = (M >> 4) * Kb;
    if (wid >= tiles) return;
    const int tm = wid / Kb, kb = wid % Kb;
    const int row   = tm * 16 + (lane & 15);
    const int kbase = kb * 32 + ((lane >> 4) ? 8 : 0);
    const float* src = A + (size_t)row * K + kbase;
    const float4 a0 = *(const float4*)(src);
    const float4 a1 = *(const float4*)(src + 4);
    const float4 b0 = *(const float4*)(src + 16);
    const float4 b1 = *(const float4*)(src + 20);
    bf16x16 f;
    f[0]=(__bf16)a0.x; f[1]=(__bf16)a0.y; f[2]=(__bf16)a0.z; f[3]=(__bf16)a0.w;
    f[4]=(__bf16)a1.x; f[5]=(__bf16)a1.y; f[6]=(__bf16)a1.z; f[7]=(__bf16)a1.w;
    f[8]=(__bf16)b0.x; f[9]=(__bf16)b0.y; f[10]=(__bf16)b0.z; f[11]=(__bf16)b0.w;
    f[12]=(__bf16)b1.x; f[13]=(__bf16)b1.y; f[14]=(__bf16)b1.z; f[15]=(__bf16)b1.w;
    *(bf16x16*)(out + ((size_t)wid * 32 + lane) * 16) = f;
}

// ---------------------------------------------------------------------------
// Repack weights W[K,N] f32 row-major -> bf16 B-fragment-major:
//   layout [tileN][kblock][lane][16]; one wave per (tileN,kblock).
// ---------------------------------------------------------------------------
__global__ __launch_bounds__(256)
void k_repack_b(const float* __restrict__ W, __bf16* __restrict__ out, int K, int N)
{
    const int lane = threadIdx.x & 31;
    const int wid  = blockIdx.x * 8 + (threadIdx.x >> 5);
    const int Kb   = K >> 5;
    const int tiles = (N >> 4) * Kb;
    if (wid >= tiles) return;
    const int tn = wid / Kb, kb = wid % Kb;
    const int col   = tn * 16 + (lane & 15);
    const int kbase = kb * 32 + ((lane >> 4) ? 8 : 0);
    bf16x16 f;
#pragma unroll
    for (int i = 0; i < 16; ++i) {
        const int k = kbase + (i < 8 ? i : i + 8);
        f[i] = (__bf16)W[(size_t)k * N + col];
    }
    *(bf16x16*)(out + ((size_t)wid * 32 + lane) * 16) = f;
}

// ---------------------------------------------------------------------------
// WMMA bf16 GEMM on repacked operands: C[M,N] = A @ W (+ bias[N]).
// Block = 8 waves; each wave computes a 32x64 slab (2 M-tiles x 4 N-tiles),
// so each B fragment read from LDS feeds 2 WMMAs. B slab (4KB/k-block) is
// double-buffered in LDS: one barrier per k-step, staging load overlaps the
// current step's WMMAs. Inner loop per 32-K: 1 staging b128/thread +
// 4 A b128 + 8 ds_load_b128 + 8 WMMA per wave.
// grid = (N/64, M/256).
// ---------------------------------------------------------------------------
__global__ __launch_bounds__(256)
void k_gemm_wmma(const __bf16* __restrict__ Abf, const __bf16* __restrict__ Bbf,
                 const float* __restrict__ bias, float* __restrict__ C,
                 int M, int N, int K)
{
    __shared__ __attribute__((aligned(32))) __bf16 Bs[2][4 * 512];  // 2 x 4KB
    const int tid  = threadIdx.x;
    const int lane = tid & 31;
    const int wave = tid >> 5;
    const int l15  = lane & 15;
    const int half = lane >> 4;
    const int Kb   = K >> 5;
    const int tm0  = blockIdx.y * 16 + wave * 2;   // two 16-row tiles per wave
    const int tn0  = blockIdx.x * 4;
    const int col0 = tn0 * 16;

    // per-thread staging source: tile (tid>>6), 16B chunk (tid&63)
    const __bf16* bsrc = Bbf + ((size_t)(tn0 + (tid >> 6)) * Kb) * 512 + (size_t)(tid & 63) * 8;
    const __bf16* a0 = Abf + (size_t)tm0 * Kb * 512 + (size_t)lane * 16;
    const __bf16* a1 = a0 + (size_t)Kb * 512;

    f32x8 acc[2][4];
#pragma unroll
    for (int t = 0; t < 4; ++t) {
        const float bv = bias ? bias[col0 + t * 16 + l15] : 0.0f;
#pragma unroll
        for (int j = 0; j < 8; ++j) { acc[0][t][j] = bv; acc[1][t][j] = bv; }
    }

    // prologue: stage k-block 0
    *(uint4*)(&Bs[0][tid * 8]) = *(const uint4*)bsrc;

    for (int kb = 0; kb < Kb; ++kb) {
        __syncthreads();                     // staged slab visible; prior reads done
        if (kb + 1 < Kb)                     // stage next slab while computing
            *(uint4*)(&Bs[(kb + 1) & 1][tid * 8]) =
                *(const uint4*)(bsrc + (size_t)(kb + 1) * 512);
        const bf16x16 af0 = *(const bf16x16*)(a0 + (size_t)kb * 512);
        const bf16x16 af1 = *(const bf16x16*)(a1 + (size_t)kb * 512);
        const __bf16* bsl = &Bs[kb & 1][lane * 16];
#pragma unroll
        for (int t = 0; t < 4; ++t) {
            const bf16x16 bf = *(const bf16x16*)(bsl + t * 512);
            acc[0][t] = __builtin_amdgcn_wmma_f32_16x16x32_bf16(
                false, af0, false, bf, (short)0, acc[0][t], false, false);
            acc[1][t] = __builtin_amdgcn_wmma_f32_16x16x32_bf16(
                false, af1, false, bf, (short)0, acc[1][t], false, false);
        }
    }

#pragma unroll
    for (int r = 0; r < 2; ++r) {
        const int mrow0 = (tm0 + r) * 16 + half * 8;
#pragma unroll
        for (int t = 0; t < 4; ++t) {
            const int col = col0 + t * 16 + l15;
#pragma unroll
            for (int j = 0; j < 8; ++j)
                C[(size_t)(mrow0 + j) * N + col] = acc[r][t][j];
        }
    }
}

// ---- broadcast bias (optionally two biases summed) over an [M,HDIM] buffer ----
__global__ void k_fill_bias(float* __restrict__ out, const float* __restrict__ b1,
                            const float* __restrict__ b2, long total)
{
    long idx = (long)blockIdx.x * blockDim.x + threadIdx.x;
    if (idx >= total) return;
    int c = (int)(idx & (HDIM - 1));
    float v = b1[c];
    if (b2) v += b2[c];
    out[idx] = v;
}

// ---- GCN edge scatter: out[dst] += ew * h[src]  (block per edge) ----
__global__ __launch_bounds__(256)
void k_gcn_scatter(const float* __restrict__ h, const int* __restrict__ src,
                   const int* __restrict__ dst, const float* __restrict__ ew,
                   float* __restrict__ out)
{
    const int e = blockIdx.x;
    const int s = src[e], d = dst[e];
    const float w = ew[e];
    const float* hr = h + (size_t)s * HDIM;
    float* od = out + (size_t)d * HDIM;
    for (int c = threadIdx.x; c < HDIM; c += blockDim.x)
        atomicAdd(&od[c], w * hr[c]);
}

// ---- rowdot: out[m] = dot(X[m,:K], v)  (one wave per row) ----
__global__ __launch_bounds__(256)
void k_rowdot(const float* __restrict__ X, const float* __restrict__ v,
              float* __restrict__ out, int Mrows, int K)
{
    const int lane = threadIdx.x & 31;
    const int row  = blockIdx.x * 8 + (threadIdx.x >> 5);
    if (row >= Mrows) return;
    const float* x = X + (size_t)row * K;
    float acc = 0.f;
    for (int k = lane; k < K; k += 32) acc += x[k] * v[k];
#pragma unroll
    for (int off = 16; off > 0; off >>= 1) acc += __shfl_down(acc, off, 32);
    if (lane == 0) out[row] = acc;
}

// ---- ordered-uint encoding for atomic float max ----
__device__ inline unsigned f2ord(float f) {
    unsigned u = __float_as_uint(f);
    return (u & 0x80000000u) ? ~u : (u | 0x80000000u);
}
__device__ inline float ord2f(unsigned u) {
    return (u & 0x80000000u) ? __uint_as_float(u & 0x7fffffffu) : __uint_as_float(~u);
}

__global__ void k_gat_init(unsigned* __restrict__ menc, float* __restrict__ sbuf, int n)
{
    int i = blockIdx.x * blockDim.x + threadIdx.x;
    if (i < n) { menc[i] = 0x007FFFFFu; /* f2ord(-inf) */ sbuf[i] = 0.f; }
}

__global__ void k_gat_logit(const float* __restrict__ as_, const float* __restrict__ ad_,
                            const int* __restrict__ src, const int* __restrict__ dst,
                            float* __restrict__ elog, unsigned* __restrict__ menc, int E)
{
    int e = blockIdx.x * blockDim.x + threadIdx.x;
    if (e >= E) return;
    float l = as_[src[e]] + ad_[dst[e]];
    l = (l > 0.f) ? l : 0.2f * l;           // leaky_relu 0.2
    elog[e] = l;
    atomicMax(&menc[dst[e]], f2ord(l));
}

__global__ void k_gat_finalm(const unsigned* __restrict__ menc, float* __restrict__ mbuf, int n)
{
    int i = blockIdx.x * blockDim.x + threadIdx.x;
    if (i >= n) return;
    float m = ord2f(menc[i]);
    mbuf[i] = isfinite(m) ? m : 0.f;        // guard empty segments
}

__global__ void k_gat_exp(float* __restrict__ elog, const float* __restrict__ mbuf,
                          const int* __restrict__ dst, float* __restrict__ sbuf, int E)
{
    int e = blockIdx.x * blockDim.x + threadIdx.x;
    if (e >= E) return;
    float ex = expf(elog[e] - mbuf[dst[e]]);
    elog[e] = ex;
    atomicAdd(&sbuf[dst[e]], ex);
}

__global__ __launch_bounds__(256)
void k_gat_scatter(const float* __restrict__ hs, const float* __restrict__ elog,
                   const float* __restrict__ sbuf, const int* __restrict__ src,
                   const int* __restrict__ dst, float* __restrict__ out)
{
    const int e = blockIdx.x;
    const int s = src[e], d = dst[e];
    const float alpha = elog[e] / sbuf[d];
    const float* hr = hs + (size_t)s * HDIM;
    float* od = out + (size_t)d * HDIM;
    for (int c = threadIdx.x; c < HDIM; c += blockDim.x)
        atomicAdd(&od[c], alpha * hr[c]);
}

// ---- fused ReLU + PyG graph-mode LayerNorm, in place. block per graph ----
__global__ __launch_bounds__(256)
void k_relu_gnorm(float* __restrict__ x, const float* __restrict__ w,
                  const float* __restrict__ b, int nper)
{
    __shared__ float s1[256], s2[256];
    const int g = blockIdx.x, tid = threadIdx.x;
    float* xg = x + (size_t)g * nper * HDIM;
    const int total = nper * HDIM;
    float a = 0.f, q = 0.f;
    for (int i = tid; i < total; i += 256) {
        float v = xg[i]; v = v > 0.f ? v : 0.f;
        a += v; q += v * v;
    }
    s1[tid] = a; s2[tid] = q; __syncthreads();
    for (int off = 128; off > 0; off >>= 1) {
        if (tid < off) { s1[tid] += s1[tid + off]; s2[tid] += s2[tid + off]; }
        __syncthreads();
    }
    const float denom = (float)total;
    const float mean  = s1[0] / denom;
    float var = s2[0] / denom - mean * mean;
    var = var > 0.f ? var : 0.f;
    const float inv = rsqrtf(var + EPSF);
    for (int i = tid; i < total; i += 256) {
        float v = xg[i]; v = v > 0.f ? v : 0.f;
        const int c = i & (HDIM - 1);
        xg[i] = w[c] * (v - mean) * inv + b[c];
    }
}

// ---- GlobalAttention readout, block per graph (audio rows contiguous) ----
__global__ __launch_bounds__(256)
void k_readout(const float* __restrict__ xa, const float* __restrict__ gate,
               float* __restrict__ embed)
{
    __shared__ float red[256];
    __shared__ float ebuf[NA_PER];
    const int g = blockIdx.x, tid = threadIdx.x;
    const int r0 = g * NA_PER;
    float lm = -INFINITY;
    for (int n = tid; n < NA_PER; n += 256) lm = fmaxf(lm, gate[r0 + n]);
    red[tid] = lm; __syncthreads();
    for (int off = 128; off > 0; off >>= 1) {
        if (tid < off) red[tid] = fmaxf(red[tid], red[tid + off]);
        __syncthreads();
    }
    float m = red[0]; if (!isfinite(m)) m = 0.f;
    __syncthreads();
    float ls = 0.f;
    for (int n = tid; n < NA_PER; n += 256) {
        float e = expf(gate[r0 + n] - m);
        ebuf[n] = e; ls += e;
    }
    red[tid] = ls; __syncthreads();
    for (int off = 128; off > 0; off >>= 1) {
        if (tid < off) red[tid] += red[tid + off];
        __syncthreads();
    }
    float s = red[0]; s = s > 1e-16f ? s : 1e-16f;
    const float invs = 1.f / s;
    for (int c = tid; c < HDIM; c += 256) {
        float acc = 0.f;
        for (int n = 0; n < NA_PER; ++n)
            acc += ebuf[n] * xa[(size_t)(r0 + n) * HDIM + c];
        embed[(size_t)g * HDIM + c] = acc * invs;
    }
}

// ---------------------------------------------------------------------------
extern "C" void kernel_launch(void* const* d_in, const int* in_sizes, int n_in,
                              void* d_out, int out_size, void* d_ws, size_t ws_size,
                              hipStream_t stream)
{
    (void)in_sizes; (void)n_in; (void)out_size; (void)ws_size;
    const float* x_video = (const float*)d_in[0];
    const float* x_audio = (const float*)d_in[1];
    const float* ew_vv   = (const float*)d_in[2];
    const float* ew_aa   = (const float*)d_in[3];
    const float* W_vv0   = (const float*)d_in[4];
    const float* W_aa0   = (const float*)d_in[5];
    const float* W_src0  = (const float*)d_in[6];
    const float* W_dst0  = (const float*)d_in[7];
    const float* W_vv    = (const float*)d_in[8];
    const float* W_aa    = (const float*)d_in[9];
    const float* W_src   = (const float*)d_in[10];
    const float* W_dst   = (const float*)d_in[11];
    const float* b_vv    = (const float*)d_in[12];
    const float* b_aa    = (const float*)d_in[13];
    const float* b_gat   = (const float*)d_in[14];
    const float* a_src   = (const float*)d_in[15];
    const float* a_dst   = (const float*)d_in[16];
    const float* ln_v_w  = (const float*)d_in[17];
    const float* ln_v_b  = (const float*)d_in[18];
    const float* ln_a_w  = (const float*)d_in[19];
    const float* ln_a_b  = (const float*)d_in[20];
    const float* att_w   = (const float*)d_in[21];
    const float* lin_W   = (const float*)d_in[22];
    const float* lin_b   = (const float*)d_in[23];
    const int*   ei_vv   = (const int*)d_in[24];
    const int*   ei_aa   = (const int*)d_in[25];
    const int*   ei_va   = (const int*)d_in[26];

    // ---- carve workspace (~515 MB) ----
    char* p = (char*)d_ws;
    auto takeb = [&](size_t bytes) -> void* {
        void* r = (void*)p;
        p += (bytes + 255) & ~(size_t)255;
        return r;
    };
    auto take = [&](size_t nfloats) -> float* { return (float*)takeb(nfloats * 4); };
    const size_t NVH = (size_t)NVN * HDIM;
    const size_t NAH = (size_t)NAN_ * HDIM;
    float* vbuf0 = take(NVH);
    float* vbuf1 = take(NVH);
    float* hv    = take(NVH);              // reused as hs
    float* abuf0 = take(NAH);
    float* abuf1 = take(NAH);
    float* ha    = take(NAH);
    __bf16* Avbf = (__bf16*)takeb((size_t)NVN * DVDIM * 2);   // max Kv = 1024
    __bf16* Aabf = (__bf16*)takeb((size_t)NAN_ * HDIM * 2);   // max Ka = 512
    __bf16* Wbf  = (__bf16*)takeb((size_t)DVDIM * HDIM * 2);  // max K*N weight
    float* as_   = take(NVN);
    float* ad_   = take(NAN_);
    float* wvec  = take(HDIM);
    float* elog  = take(EVA);
    unsigned* menc = (unsigned*)take(NAN_);
    float* mbuf  = take(NAN_);
    float* sbuf  = take(NAN_);
    float* gate  = take(NAN_);
    float* embed = take((size_t)B_GRAPHS * HDIM);

    float* xv_cur = nullptr;
    float* xa_cur = nullptr;

    const dim3 gv(HDIM / 64, NVN / 256);
    const dim3 ga(HDIM / 64, NAN_ / 256);
    auto rgA = [](int M, int K) { return (unsigned)(((M >> 4) * (K >> 5) + 7) / 8); };
    auto rgB = [](int K, int N) { return (unsigned)(((N >> 4) * (K >> 5) + 7) / 8); };

    for (int i = 0; i < LAYERS; ++i) {
        const float* xv_in = i ? xv_cur : x_video;
        const float* xa_in = i ? xa_cur : x_audio;
        const int Kv = i ? HDIM : DVDIM;
        const int Ka = i ? HDIM : DADIM;
        float* aggv = (i & 1) ? vbuf1 : vbuf0;
        float* agga = (i & 1) ? abuf1 : abuf0;
        const size_t wo = (size_t)(i ? (i - 1) : 0) * HDIM * HDIM;
        const float* Wv = i ? (W_vv  + wo) : W_vv0;
        const float* Wa = i ? (W_aa  + wo) : W_aa0;
        const float* Ws = i ? (W_src + wo) : W_src0;
        const float* Wd = i ? (W_dst + wo) : W_dst0;

        // video GCN: hv = xv @ Wv ; aggv = b_vv + scatter(ew * hv[src])
        k_repack_a<<<rgA(NVN, Kv), 256, 0, stream>>>(xv_in, Avbf, NVN, Kv);
        k_repack_b<<<rgB(Kv, HDIM), 256, 0, stream>>>(Wv, Wbf, Kv, HDIM);
        k_gemm_wmma<<<gv, 256, 0, stream>>>(Avbf, Wbf, nullptr, hv, NVN, HDIM, Kv);
        k_fill_bias<<<(unsigned)((NVH + 255) / 256), 256, 0, stream>>>(
            aggv, b_vv + i * HDIM, nullptr, (long)NVH);
        k_gcn_scatter<<<EVV, 256, 0, stream>>>(hv, ei_vv, ei_vv + EVV, ew_vv, aggv);

        // audio GCN: ha = xa @ Wa ; agga = b_aa + b_gat + scatter(ew * ha[src])
        k_repack_a<<<rgA(NAN_, Ka), 256, 0, stream>>>(xa_in, Aabf, NAN_, Ka);
        k_repack_b<<<rgB(Ka, HDIM), 256, 0, stream>>>(Wa, Wbf, Ka, HDIM);
        k_gemm_wmma<<<ga, 256, 0, stream>>>(Aabf, Wbf, nullptr, ha, NAN_, HDIM, Ka);
        k_fill_bias<<<(unsigned)((NAH + 255) / 256), 256, 0, stream>>>(
            agga, b_aa + i * HDIM, b_gat + i * HDIM, (long)NAH);
        k_gcn_scatter<<<EAA, 256, 0, stream>>>(ha, ei_aa, ei_aa + EAA, ew_aa, agga);

        // GAT video->audio: hs = xv @ Ws (reuses repacked xv); ad = xa @ (Wd @ a_d)
        k_repack_b<<<rgB(Kv, HDIM), 256, 0, stream>>>(Ws, Wbf, Kv, HDIM);
        k_gemm_wmma<<<gv, 256, 0, stream>>>(Avbf, Wbf, nullptr, hv, NVN, HDIM, Kv);
        k_rowdot<<<(NVN + 7) / 8, 256, 0, stream>>>(hv, a_src + i * HDIM, as_, NVN, HDIM);
        k_rowdot<<<(Ka + 7) / 8, 256, 0, stream>>>(Wd, a_dst + i * HDIM, wvec, Ka, HDIM);
        k_rowdot<<<(NAN_ + 7) / 8, 256, 0, stream>>>(xa_in, wvec, ad_, NAN_, Ka);
        k_gat_init<<<(NAN_ + 255) / 256, 256, 0, stream>>>(menc, sbuf, NAN_);
        k_gat_logit<<<(EVA + 255) / 256, 256, 0, stream>>>(
            as_, ad_, ei_va, ei_va + EVA, elog, menc, EVA);
        k_gat_finalm<<<(NAN_ + 255) / 256, 256, 0, stream>>>(menc, mbuf, NAN_);
        k_gat_exp<<<(EVA + 255) / 256, 256, 0, stream>>>(elog, mbuf, ei_va + EVA, sbuf, EVA);
        k_gat_scatter<<<EVA, 256, 0, stream>>>(hv, elog, sbuf, ei_va, ei_va + EVA, agga);

        // ReLU + graph norm (in place)
        k_relu_gnorm<<<B_GRAPHS, 256, 0, stream>>>(
            aggv, ln_v_w + i * HDIM, ln_v_b + i * HDIM, NV_PER);
        k_relu_gnorm<<<B_GRAPHS, 256, 0, stream>>>(
            agga, ln_a_w + i * HDIM, ln_a_b + i * HDIM, NA_PER);

        xv_cur = aggv;
        xa_cur = agga;
    }

    // readout + final linear
    k_rowdot<<<(NAN_ + 7) / 8, 256, 0, stream>>>(xa_cur, att_w, gate, NAN_, HDIM);
    k_readout<<<B_GRAPHS, 256, 0, stream>>>(xa_cur, gate, embed);
    k_repack_a<<<rgA(B_GRAPHS, HDIM), 256, 0, stream>>>(embed, Avbf, B_GRAPHS, HDIM);
    k_repack_b<<<rgB(HDIM, OUTD), 256, 0, stream>>>(lin_W, Wbf, HDIM, OUTD);
    const dim3 go(OUTD / 64, B_GRAPHS / 256);
    k_gemm_wmma<<<go, 256, 0, stream>>>(Avbf, Wbf, lin_b, (float*)d_out,
                                        B_GRAPHS, OUTD, HDIM);
}
